// encoder_25477746000222
// MI455X (gfx1250) — matrix operand order repaired and checked by reading, not compiled, not run on hardware
//
#include <hip/hip_runtime.h>
#include <hip/hip_bf16.h>
#include <math.h>

// ---------------------------------------------------------------------------
// CDNA5 (gfx1250) implementation. wave32. Matrix ops via
// v_wmma_f32_16x16x32_bf16; TDM async prefetch of the next A panel.
// ---------------------------------------------------------------------------

typedef __attribute__((ext_vector_type(16))) __bf16 v16bf;
typedef __attribute__((ext_vector_type(8)))  float  v8f;

#if defined(__has_builtin)
#if __has_builtin(__builtin_amdgcn_tensor_load_to_lds) && \
    __has_builtin(__builtin_amdgcn_s_wait_tensorcnt)
#define HAVE_TDM 1
#endif
#endif
#ifndef HAVE_TDM
#define HAVE_TDM 0
#endif

#if HAVE_TDM
typedef __attribute__((ext_vector_type(4))) unsigned int u32x4;
typedef __attribute__((ext_vector_type(8))) int          i32x8;
typedef __attribute__((ext_vector_type(4))) int          i32x4;
#endif

__device__ __forceinline__ unsigned short f2bf(float f) {
  union { float f; unsigned int u; } x; x.f = f;
  unsigned int u = x.u;
  unsigned int r = u + 0x7fffu + ((u >> 16) & 1u);   // round-to-nearest-even
  if ((u & 0x7f800000u) == 0x7f800000u) r = u;       // inf/nan passthrough
  return (unsigned short)(r >> 16);
}
__device__ __forceinline__ unsigned int pack_bf2(float lo, float hi) {
  return (unsigned int)f2bf(lo) | ((unsigned int)f2bf(hi) << 16);
}
__device__ __forceinline__ float silu_f(float x) { return x / (1.f + __expf(-x)); }

// ---------------------------------------------------------------------------
// Generic GEMM:  C[M,N] = A[M,K] * B + bias_n[n] + bias_m[m]
//   bkn == 0 : B is (N,K) row-major (y = A @ W^T), ldb = K-stride
//   bkn == 1 : B is (K,N) row-major, ldb = N-stride (transposed in staging)
// Block: 256 threads = 8 waves; tile 64(M) x 64(N); K step 32.
// Waves: 4(M) x 2(N); each wave does a 16x32 tile = 2 WMMA sharing one A frag.
// All leading dims in this model are even -> float2 fast path is aligned.
// ---------------------------------------------------------------------------
#define BM 64
#define BN 64
#define BK 32
#define LDK 34   // LDS row stride in bf16 (even -> dword-aligned pair reads)

__global__ __launch_bounds__(256) void gemm_wmma_kernel(
    const float* __restrict__ A, int lda,
    const float* __restrict__ Bg, int ldb, int bkn,
    const float* __restrict__ bias_n, const float* __restrict__ bias_m,
    float* __restrict__ C, int ldc, int M, int N, int K)
{
  __shared__ __align__(16) unsigned short As[BM * LDK];
  __shared__ __align__(16) unsigned short Bs[BN * LDK];
#if HAVE_TDM
  __shared__ __align__(16) float tdm_buf[BM * BK];   // async-prefetch landing pad
#endif

  const int tid   = threadIdx.x;
  const int lane  = tid & 31;
  const int wid   = tid >> 5;
  const int wm    = wid & 3;        // 0..3 (M tiles)
  const int wn    = wid >> 2;       // 0..1 (N tiles of 32)
  const int m0    = blockIdx.y * BM;
  const int n0    = blockIdx.x * BN;
  const int half  = lane >> 4;
  const int row16 = lane & 15;

  v8f acc0 = {};
  v8f acc1 = {};

  for (int kt = 0; kt < K; kt += BK) {
    const bool full = (m0 + BM <= M) && (n0 + BN <= N) && (kt + BK <= K);

    if (full) {
      // ---- fast path: float2 loads, packed ds_store_b32 ----
#pragma unroll
      for (int j = 0; j < 4; ++j) {                 // A: 64x32
        int e = tid + 256 * j;
        int r = e >> 4, c2 = e & 15;
        float2 v = *(const float2*)(A + (long)(m0 + r) * lda + kt + c2 * 2);
        *(unsigned int*)&As[r * LDK + c2 * 2] = pack_bf2(v.x, v.y);
      }
      if (bkn == 0) {
#pragma unroll
        for (int j = 0; j < 4; ++j) {               // B: 64n x 32k
          int e = tid + 256 * j;
          int r = e >> 4, c2 = e & 15;
          float2 v = *(const float2*)(Bg + (long)(n0 + r) * ldb + kt + c2 * 2);
          *(unsigned int*)&Bs[r * LDK + c2 * 2] = pack_bf2(v.x, v.y);
        }
      } else {
#pragma unroll
        for (int j = 0; j < 4; ++j) {               // B: 32k x 64n, transpose
          int e = tid + 256 * j;
          int k = e >> 5, n2 = e & 31;
          float2 v = *(const float2*)(Bg + (long)(kt + k) * ldb + n0 + n2 * 2);
          Bs[(n2 * 2 + 0) * LDK + k] = f2bf(v.x);
          Bs[(n2 * 2 + 1) * LDK + k] = f2bf(v.y);
        }
      }
    } else {
      // ---- slow path: guarded scalar loads (edge tiles only) ----
#pragma unroll
      for (int j = 0; j < 8; ++j) {                 // A: 64x32
        int e = tid + 256 * j;
        int r = e >> 5, k = e & 31;
        int gm = m0 + r, gk = kt + k;
        float v = (gm < M && gk < K) ? A[(long)gm * lda + gk] : 0.f;
        As[r * LDK + k] = f2bf(v);
      }
      if (bkn == 0) {
#pragma unroll
        for (int j = 0; j < 8; ++j) {               // B: 64n x 32k
          int e = tid + 256 * j;
          int r = e >> 5, k = e & 31;
          int gn = n0 + r, gk = kt + k;
          float v = (gn < N && gk < K) ? Bg[(long)gn * ldb + gk] : 0.f;
          Bs[r * LDK + k] = f2bf(v);
        }
      } else {
#pragma unroll
        for (int j = 0; j < 8; ++j) {               // B: 32k x 64n
          int e = tid + 256 * j;
          int k = e >> 6, n = e & 63;
          int gn = n0 + n, gk = kt + k;
          float v = (gn < N && gk < K) ? Bg[(long)gk * ldb + gn] : 0.f;
          Bs[n * LDK + k] = f2bf(v);
        }
      }
    }
    __syncthreads();

#if HAVE_TDM
    // Async TDM prefetch of the NEXT 64x32 A panel into scratch LDS.
    // Pure L2/LDS warm-up: computed results never read tdm_buf, so a
    // descriptor-layout mistake cannot corrupt the output.
    if (wid == 0 && (kt + 2 * BK <= K) && (m0 + BM <= M)) {
      unsigned long long ga =
          (unsigned long long)(const void*)(A + (long)m0 * lda + kt + BK);
      unsigned int lds_off = (unsigned int)(unsigned long long)(const void*)tdm_buf;
      u32x4 g0 = { 1u,                                   // count=1, user D#
                   lds_off,                              // lds_addr
                   (unsigned int)(ga & 0xffffffffu),     // global_addr[31:0]
                   (unsigned int)((ga >> 32) & 0x01ffffffu) | (2u << 30) }; // +type=2
      i32x8 g1 = { (int)(2u << 16),                      // data_size=4B
                   (int)(32u << 16),                     // tensor_dim0 = 32
                   (int)(64u << 16),                     // tensor_dim1 = 64
                   (int)(32u << 16),                     // tile_dim0   = 32
                   (int)64u,                             // tile_dim1   = 64
                   lda,                                  // tensor_dim0_stride
                   0, 0 };
      i32x4 gz4 = { 0, 0, 0, 0 };
      i32x8 gz8 = { 0, 0, 0, 0, 0, 0, 0, 0 };
      __builtin_amdgcn_tensor_load_to_lds(g0, g1, gz4, gz4, gz8, 0);
    }
#else
    if (kt + BK < K) {                                   // global_prefetch_b8
      int r = tid >> 5, k = tid & 31;
      long gm = m0 + r;
      if (gm < M) __builtin_prefetch(A + gm * lda + kt + BK + k, 0, 1);
    }
#endif

    // ---- fragments in contiguous k order (k = 16*half + s); one A frag feeds two B frags ----
    union { v16bf v; unsigned int u[8]; } av, bv0, bv1;
    const unsigned short* ap  = &As[(wm * 16 + row16) * LDK];
    const unsigned short* bp0 = &Bs[(wn * 32 + row16) * LDK];
    const unsigned short* bp1 = &Bs[(wn * 32 + 16 + row16) * LDK];
#pragma unroll
    for (int r = 0; r < 8; ++r) {
      int k0a = (r << 1) + (half << 4);                          // A 16x32
      int k0b = (r << 1) + (half << 4);                          // B 32x16
      av.u[r]  = *(const unsigned int*)(ap  + k0a);
      bv0.u[r] = *(const unsigned int*)(bp0 + k0b);
      bv1.u[r] = *(const unsigned int*)(bp1 + k0b);
    }
    acc0 = __builtin_amdgcn_wmma_f32_16x16x32_bf16(
        false, av.v, false, bv0.v, (short)0, acc0, false, false);
    acc1 = __builtin_amdgcn_wmma_f32_16x16x32_bf16(
        false, av.v, false, bv1.v, (short)0, acc1, false, false);
    __syncthreads();
  }

#if HAVE_TDM
  __builtin_amdgcn_s_wait_tensorcnt(0);
#endif

  // ---- store: C VGPR r -> M = r + 8*half, N = lane&15 ----
#pragma unroll
  for (int r = 0; r < 8; ++r) {
    int gm  = m0 + wm * 16 + r + half * 8;
    int gn0 = n0 + wn * 32 + row16;
    int gn1 = gn0 + 16;
    if (gm < M) {
      float bm_v = bias_m ? bias_m[gm] : 0.f;
      if (gn0 < N) {
        float v = acc0[r] + bm_v;
        if (bias_n) v += bias_n[gn0];
        C[(long)gm * ldc + gn0] = v;
      }
      if (gn1 < N) {
        float v = acc1[r] + bm_v;
        if (bias_n) v += bias_n[gn1];
        C[(long)gm * ldc + gn1] = v;
      }
    }
  }
}

// ---------------------------------------------------------------------------
// conv3x3 stride2 pad1 + BatchNorm + exact GELU   (NCHW)
// ---------------------------------------------------------------------------
__global__ void conv_bn_gelu_kernel(
    const float* __restrict__ x, const float* __restrict__ w,
    const float* __restrict__ cb, const float* __restrict__ g,
    const float* __restrict__ bb, const float* __restrict__ bm,
    const float* __restrict__ bv, float* __restrict__ y,
    int B, int Cin, int Hin, int Win, int Co, int Ho, int Wo)
{
  long idx = (long)blockIdx.x * blockDim.x + threadIdx.x;
  long total = (long)B * Co * Ho * Wo;
  if (idx >= total) return;
  int wo = idx % Wo; long t = idx / Wo;
  int ho = t % Ho; t /= Ho;
  int co = t % Co; int b = t / Co;

  float acc = cb[co];
  for (int ci = 0; ci < Cin; ++ci) {
    const float* xp = x + ((long)(b * Cin + ci)) * Hin * Win;
    const float* wp = w + ((long)(co * Cin + ci)) * 9;
#pragma unroll
    for (int r = 0; r < 3; ++r) {
      int ih = ho * 2 - 1 + r;
      if (ih < 0 || ih >= Hin) continue;
#pragma unroll
      for (int s = 0; s < 3; ++s) {
        int iw = wo * 2 - 1 + s;
        if (iw < 0 || iw >= Win) continue;
        acc += wp[r * 3 + s] * xp[(long)ih * Win + iw];
      }
    }
  }
  float v = (acc - bm[co]) * rsqrtf(bv[co] + 1e-5f) * g[co] + bb[co];
  y[idx] = 0.5f * v * (1.f + erff(v * 0.70710678118654752f));
}

// ---------------------------------------------------------------------------
// layout shuffles
// ---------------------------------------------------------------------------
__global__ void nchw_to_nhwc_kernel(const float* __restrict__ src,
                                    float* __restrict__ dst,
                                    int B, int C, int HW)
{
  long idx = (long)blockIdx.x * blockDim.x + threadIdx.x;
  long total = (long)B * C * HW;
  if (idx >= total) return;
  int hw = idx % HW; long t = idx / HW;
  int c = t % C; int b = t / C;
  dst[((long)b * HW + hw) * C + c] = src[idx];
}

__global__ void nhwc_to_nchw_cat_kernel(const float* __restrict__ src,
                                        float* __restrict__ dst,
                                        int B, int C, int HW,
                                        int c_off, int Ctot)
{
  long idx = (long)blockIdx.x * blockDim.x + threadIdx.x;
  long total = (long)B * HW * C;
  if (idx >= total) return;
  int c = idx % C; long t = idx / C;
  int hw = t % HW; int b = t / HW;
  dst[((long)b * Ctot + c_off + c) * HW + hw] = src[idx];
}

// ---------------------------------------------------------------------------
// 3x3 neighborhood attention (NATTEN), qkv buffer is (B*HW, 3*C)
// ---------------------------------------------------------------------------
__global__ void natten_kernel(const float* __restrict__ qkv,
                              const float* __restrict__ rpb,
                              float* __restrict__ out,
                              int B, int H, int W, int heads, int hd)
{
  const int C = heads * hd;
  long idx = (long)blockIdx.x * blockDim.x + threadIdx.x;
  long total = (long)B * H * W * heads;
  if (idx >= total) return;
  int hh = idx % heads; long t = idx / heads;
  int w = t % W; t /= W;
  int h = t % H; int b = t / H;
  long tok = ((long)b * H + h) * W + w;

  float q[32];
  const float scale = rsqrtf((float)hd);
  const float* qp = qkv + tok * 3 * C + hh * hd;
  for (int e = 0; e < hd; ++e) q[e] = qp[e] * scale;

  int si = min(max(h - 1, 0), H - 3);
  int sj = min(max(w - 1, 0), W - 3);

  float logit[9];
#pragma unroll
  for (int a = 0; a < 3; ++a)
#pragma unroll
    for (int c2 = 0; c2 < 3; ++c2) {
      int ni = si + a, nj = sj + c2;
      long ntok = ((long)b * H + ni) * W + nj;
      const float* kp = qkv + ntok * 3 * C + C + hh * hd;
      float dot = 0.f;
      for (int e = 0; e < hd; ++e) dot += q[e] * kp[e];
      int ri = ni - h + 2, rj = nj - w + 2;
      logit[a * 3 + c2] = dot + rpb[(hh * 5 + ri) * 5 + rj];
    }

  float mx = logit[0];
#pragma unroll
  for (int i = 1; i < 9; ++i) mx = fmaxf(mx, logit[i]);
  float se = 0.f;
#pragma unroll
  for (int i = 0; i < 9; ++i) { logit[i] = __expf(logit[i] - mx); se += logit[i]; }
  float inv = 1.f / se;

  float o[32];
  for (int e = 0; e < hd; ++e) o[e] = 0.f;
#pragma unroll
  for (int a = 0; a < 3; ++a)
#pragma unroll
    for (int c2 = 0; c2 < 3; ++c2) {
      int ni = si + a, nj = sj + c2;
      long ntok = ((long)b * H + ni) * W + nj;
      const float* vp = qkv + ntok * 3 * C + 2 * C + hh * hd;
      float wt = logit[a * 3 + c2] * inv;
      for (int e = 0; e < hd; ++e) o[e] += wt * vp[e];
    }
  float* op = out + tok * C + hh * hd;
  for (int e = 0; e < hd; ++e) op[e] = o[e];
}

// ---------------------------------------------------------------------------
// Mamba causal depthwise conv (D_CONV=4) + SiLU. xz is (B*L, 2*di), xm half.
// ---------------------------------------------------------------------------
__global__ void mamba_conv_silu_kernel(const float* __restrict__ xz,
                                       const float* __restrict__ cw,
                                       const float* __restrict__ cb,
                                       float* __restrict__ xc,
                                       int B, int L, int di)
{
  long idx = (long)blockIdx.x * blockDim.x + threadIdx.x;
  long total = (long)B * L * di;
  if (idx >= total) return;
  int d = idx % di; long t = idx / di;
  int l = t % L; int b = t / L;
  float acc = cb[d];
#pragma unroll
  for (int j = 0; j < 4; ++j) {
    int ls = l - 3 + j;
    if (ls >= 0) acc += cw[d * 4 + j] * xz[((long)b * L + ls) * 2 * di + d];
  }
  xc[idx] = silu_f(acc);
}

// ---------------------------------------------------------------------------
// Selective scan: h = h*exp(dt*A) + dt*xc*B ; y = <h,C> ; gate with silu(z).
// One thread per (b, d) channel; 16 states in registers.
// ---------------------------------------------------------------------------
__global__ void mamba_scan_kernel(const float* __restrict__ dtraw,
                                  const float* __restrict__ dbc, int ldd, int dtr,
                                  const float* __restrict__ xc,
                                  const float* __restrict__ xz,
                                  const float* __restrict__ A_log,
                                  const float* __restrict__ Dp,
                                  float* __restrict__ y,
                                  int B, int L, int di)
{
  int idx = blockIdx.x * blockDim.x + threadIdx.x;
  if (idx >= B * di) return;
  int d = idx % di, b = idx / di;

  float Arow[16], h[16];
#pragma unroll
  for (int s = 0; s < 16; ++s) { Arow[s] = -__expf(A_log[d * 16 + s]); h[s] = 0.f; }
  const float Dd = Dp[d];
  const long base = (long)b * L;

  for (int l = 0; l < L; ++l) {
    long t = base + l;
    float dtv = dtraw[t * di + d];
    dtv = (dtv > 20.f) ? dtv : log1pf(__expf(dtv));          // softplus
    float xcv = xc[t * di + d];
    float dx = dtv * xcv;
    const float* db = dbc + t * ldd + dtr;                    // B then C
    float acc = 0.f;
#pragma unroll
    for (int s = 0; s < 16; ++s) {
      float dA = __expf(dtv * Arow[s]);
      h[s] = h[s] * dA + dx * db[s];
      acc += h[s] * db[16 + s];
    }
    float zv = xz[t * 2 * di + di + d];
    y[t * di + d] = (acc + xcv * Dd) * silu_f(zv);
  }
}

// ---------------------------------------------------------------------------
// RMSNorm over channel dim
// ---------------------------------------------------------------------------
__global__ void rmsnorm_kernel(const float* __restrict__ x,
                               const float* __restrict__ w,
                               float* __restrict__ out, long T, int C)
{
  long t = (long)blockIdx.x * blockDim.x + threadIdx.x;
  if (t >= T) return;
  const float* xp = x + t * C;
  float ss = 0.f;
  for (int c = 0; c < C; ++c) { float v = xp[c]; ss += v * v; }
  float r = rsqrtf(ss / (float)C + 1e-5f);
  float* op = out + t * C;
  for (int c = 0; c < C; ++c) op[c] = xp[c] * r * w[c];
}

// ---------------------------------------------------------------------------
// Host orchestration
// ---------------------------------------------------------------------------
namespace {

struct MambaP {
  const float *A_log, *D, *conv_b, *conv_w, *dt_proj_b, *dt_proj_w,
              *in_proj_w, *out_proj_w, *x_proj_w;
};
struct StageP {
  const float *bn_b, *bn_g, *bn_m, *bn_v, *c1_b, *c1_w, *conv_b, *conv_w;
  MambaP m[6];
  const float *proj_b, *proj_w, *qkv_b, *qkv_w, *rms_w, *rpb;
};

inline void launch_gemm(hipStream_t st, const float* A, int lda,
                        const float* Bg, int ldb, int bkn,
                        const float* bias_n, const float* bias_m,
                        float* C, int ldc, int M, int N, int K)
{
  dim3 grid((N + BN - 1) / BN, (M + BM - 1) / BM);
  gemm_wmma_kernel<<<grid, 256, 0, st>>>(A, lda, Bg, ldb, bkn,
                                         bias_n, bias_m, C, ldc, M, N, K);
}

inline int g1d(long n) { return (int)((n + 255) / 256); }

} // namespace

extern "C" void kernel_launch(void* const* d_in, const int* in_sizes, int n_in,
                              void* d_out, int out_size, void* d_ws, size_t ws_size,
                              hipStream_t stream)
{
  // ---- workspace carve-out (floats), sized by stage-1 maxima ----
  const size_t O_ACT  = 0;                       // conv out NCHW
  const size_t O_NHWC = O_ACT  + 1048576;        // tokens (BL, C)
  const size_t O_QKV  = O_NHWC + 1048576;        // (BL, 3C)
  const size_t O_ATTN = O_QKV  + 3145728;        // (BL, C)
  const size_t O_TOK0 = O_ATTN + 1048576;
  const size_t O_TOK1 = O_TOK0 + 1048576;
  const size_t O_XZ   = O_TOK1 + 1048576;        // (BL, 2di)
  const size_t O_XC   = O_XZ   + 4194304;        // (BL, di)
  const size_t O_DBC  = O_XC   + 2097152;        // (BL, dtr+32)
  const size_t O_DT   = O_DBC  + 1572864;        // (BL, di)
  const size_t O_Y    = O_DT   + 2097152;        // (BL, di)
  const size_t O_CAT  = O_Y    + 2097152;        // (B, 2C, HW)
  const size_t WS_FLOATS = O_CAT + 2097152;
  if (ws_size < WS_FLOATS * sizeof(float)) return;

  float* ws = (float*)d_ws;
  float* act  = ws + O_ACT;   float* nhwc = ws + O_NHWC;
  float* qkvb = ws + O_QKV;   float* attn = ws + O_ATTN;
  float* tok0 = ws + O_TOK0;  float* tok1 = ws + O_TOK1;
  float* xz   = ws + O_XZ;    float* xc   = ws + O_XC;
  float* dbc  = ws + O_DBC;   float* dtb  = ws + O_DT;
  float* yb   = ws + O_Y;     float* cat  = ws + O_CAT;
  float* outp = (float*)d_out;

  // ---- param pointers: 'x' first, then params with dict keys sorted ----
  int pi = 0;
  const float* x_in = (const float*)d_in[pi++];
  StageP sp[4];
  for (int s = 0; s < 4; ++s) {
    StageP& p = sp[s];
    p.bn_b   = (const float*)d_in[pi++]; p.bn_g   = (const float*)d_in[pi++];
    p.bn_m   = (const float*)d_in[pi++]; p.bn_v   = (const float*)d_in[pi++];
    p.c1_b   = (const float*)d_in[pi++]; p.c1_w   = (const float*)d_in[pi++];
    p.conv_b = (const float*)d_in[pi++]; p.conv_w = (const float*)d_in[pi++];
    for (int i = 0; i < 6; ++i) {
      MambaP& mm = p.m[i];
      mm.A_log     = (const float*)d_in[pi++]; mm.D         = (const float*)d_in[pi++];
      mm.conv_b    = (const float*)d_in[pi++]; mm.conv_w    = (const float*)d_in[pi++];
      mm.dt_proj_b = (const float*)d_in[pi++]; mm.dt_proj_w = (const float*)d_in[pi++];
      mm.in_proj_w = (const float*)d_in[pi++]; mm.out_proj_w= (const float*)d_in[pi++];
      mm.x_proj_w  = (const float*)d_in[pi++];
    }
    p.proj_b = (const float*)d_in[pi++]; p.proj_w = (const float*)d_in[pi++];
    p.qkv_b  = (const float*)d_in[pi++]; p.qkv_w  = (const float*)d_in[pi++];
    p.rms_w  = (const float*)d_in[pi++]; p.rpb    = (const float*)d_in[pi++];
  }

  const int B = 2;
  const int cins[4]  = {3, 64, 128, 256};
  const int dims[4]  = {32, 64, 128, 256};
  const int headsA[4]= {4, 4, 8, 8};
  const int hins[4]  = {256, 128, 64, 32};
  const long out_off[4] = {0, 2097152, 3145728, 3670016};

  const float* stage_in = x_in;
  for (int s = 0; s < 4; ++s) {
    const StageP& p = sp[s];
    const int Cin = cins[s], C = dims[s], heads = headsA[s];
    const int Hin = hins[s], Win = hins[s];
    const int H = Hin / 2, W = Win / 2, HW = H * W;
    const int BL = B * HW;
    const int di = 2 * C;
    const int dtr = (C + 15) / 16;
    const int ldd = dtr + 32;
    const int hd = C / heads;

    // 1) conv3x3/s2 + BN + GELU -> act (NCHW)
    conv_bn_gelu_kernel<<<g1d((long)B * C * HW), 256, 0, stream>>>(
        stage_in, p.conv_w, p.conv_b, p.bn_g, p.bn_b, p.bn_m, p.bn_v,
        act, B, Cin, Hin, Win, C, H, W);

    // 2) NCHW -> tokens (BL, C)
    nchw_to_nhwc_kernel<<<g1d((long)B * C * HW), 256, 0, stream>>>(
        act, nhwc, B, C, HW);

    // 3) qkv = tokens @ qkv_w^T + qkv_b    (WMMA)
    launch_gemm(stream, nhwc, C, p.qkv_w, C, 0, p.qkv_b, nullptr,
                qkvb, 3 * C, BL, 3 * C, C);

    // 4) 3x3 neighborhood attention
    natten_kernel<<<g1d((long)B * HW * heads), 256, 0, stream>>>(
        qkvb, p.rpb, attn, B, H, W, heads, hd);

    // 5) proj -> tok0 (also the Mamba token input)
    launch_gemm(stream, attn, C, p.proj_w, C, 0, p.proj_b, nullptr,
                tok0, C, BL, C, C);

    // 6) xa (NCHW) into first half of concat buffer
    nhwc_to_nchw_cat_kernel<<<g1d((long)BL * C), 256, 0, stream>>>(
        tok0, cat, B, C, HW, 0, 2 * C);

    // 7) six Mamba blocks (ping-pong tok0/tok1)
    float* tin = tok0; float* tout = tok1;
    for (int i = 0; i < 6; ++i) {
      const MambaP& mm = p.m[i];
      launch_gemm(stream, tin, C, mm.in_proj_w, C, 0, nullptr, nullptr,
                  xz, 2 * di, BL, 2 * di, C);
      mamba_conv_silu_kernel<<<g1d((long)BL * di), 256, 0, stream>>>(
          xz, mm.conv_w, mm.conv_b, xc, B, HW, di);
      launch_gemm(stream, xc, di, mm.x_proj_w, di, 0, nullptr, nullptr,
                  dbc, ldd, BL, ldd, di);
      launch_gemm(stream, dbc, ldd, mm.dt_proj_w, dtr, 0, mm.dt_proj_b, nullptr,
                  dtb, di, BL, di, dtr);
      mamba_scan_kernel<<<(B * di + 127) / 128, 128, 0, stream>>>(
          dtb, dbc, ldd, dtr, xc, xz, mm.A_log, mm.D, yb, B, HW, di);
      launch_gemm(stream, yb, di, mm.out_proj_w, di, 0, nullptr, nullptr,
                  tout, C, BL, C, di);
      float* tmp = tin; tin = tout; tout = tmp;
    }
    // after even # of swaps result is in tok0
    rmsnorm_kernel<<<g1d(BL), 256, 0, stream>>>(tok0, p.rms_w, tok1, BL, C);

    // 8) xm (NCHW) into second half of concat buffer
    nhwc_to_nchw_cat_kernel<<<g1d((long)BL * C), 256, 0, stream>>>(
        tok1, cat, B, C, HW, C, 2 * C);

    // 9) 1x1 conv: per batch, out[o,hw] = c1_w[o,:] @ cat[b,:,hw] + c1_b[o]
    float* stage_out = outp + out_off[s];
    for (int b = 0; b < B; ++b) {
      launch_gemm(stream, p.c1_w, 2 * C,
                  cat + (long)b * 2 * C * HW, HW, /*bkn=*/1,
                  nullptr, p.c1_b,
                  stage_out + (long)b * 2 * C * HW, HW,
                  2 * C, HW, 2 * C);
    }
    stage_in = stage_out;   // next stage consumes this stage's output
  }
  (void)in_sizes; (void)n_in; (void)out_size;
}
